// SceneGraphGAT_28200755265596
// MI455X (gfx1250) — matrix operand order repaired and checked by reading, not compiled, not run on hardware
//
#include <hip/hip_runtime.h>
#include <hip/hip_bf16.h>
#include <math.h>

typedef __attribute__((ext_vector_type(16))) __bf16 v16bf;
typedef __attribute__((ext_vector_type(8)))  __bf16 v8bf;
typedef __attribute__((ext_vector_type(8)))  float  v8f;

#define DF 128

// ---------------- utility kernels ----------------

__global__ void __launch_bounds__(256)
cvt_f32_to_bf16(const float* __restrict__ in, __bf16* __restrict__ out, int n)
{
    int i = blockIdx.x * blockDim.x + threadIdx.x;
    if (i < n) out[i] = (__bf16)in[i];
}

__global__ void __launch_bounds__(256)
fill_f32(float* __restrict__ p, float v, int n)
{
    int i = blockIdx.x * blockDim.x + threadIdx.x;
    if (i < n) p[i] = v;
}

__global__ void __launch_bounds__(256)
elu_to_bf16(const float* __restrict__ h, __bf16* __restrict__ out, int n)
{
    int i = blockIdx.x * blockDim.x + threadIdx.x;
    if (i < n) {
        float x = h[i];
        float y = x > 0.0f ? x : (__expf(x) - 1.0f);
        out[i] = (__bf16)y;
    }
}

// ---------------- fused Q/K/V projection via WMMA bf16 ----------------
// Block: 16 node rows x 256 threads (8 waves). Wave w owns output columns
// [16w,16w+16) and computes that tile for Q, K and V (A-fragments reused).
// D = A(16x128 bf16) x W^T(128x16 bf16) + bias, f32 accumulate.
__global__ void __launch_bounds__(256)
gemm_qkv_bf16(const __bf16* __restrict__ X,
              const __bf16* __restrict__ Wq, const __bf16* __restrict__ Wk,
              const __bf16* __restrict__ Wv,
              const float* __restrict__ bq, const float* __restrict__ bk,
              const float* __restrict__ bv,
              float* __restrict__ Q, float* __restrict__ K, float* __restrict__ V,
              int nrows)
{
    const int lane = threadIdx.x & 31;
    const int wv   = threadIdx.x >> 5;    // 0..7 : output column tile
    const int r0   = blockIdx.x * 16;
    const int c0   = wv * 16;
    const int nsub = lane & 15;
    const int half = lane >> 4;           // 0 or 1
    const bool full_tile = (r0 + 16) <= nrows;   // wave-uniform (scalar branch)

    // A fragments (16x32 bf16 per k-step), contiguous k order (same as B):
    // lane<16: M=lane,    K = kb+{0..15}
    // lane>=16:M=lane-16, K = kb+16+{0..15}
    int arow = r0 + nsub; if (arow >= nrows) arow = nrows - 1;   // clamp tail
    const __bf16* xrow = X + (size_t)arow * DF;
    v16bf a[4];
#pragma unroll
    for (int ks = 0; ks < 4; ++ks) {
        const int kb = ks * 32 + half * 16;
        v8bf lo = *(const v8bf*)(xrow + kb);
        v8bf hi = *(const v8bf*)(xrow + kb + 8);
#pragma unroll
        for (int j = 0; j < 8; ++j) { a[ks][j] = lo[j]; a[ks][8 + j] = hi[j]; }
    }

    const __bf16* Ws[3] = {Wq, Wk, Wv};
    const float*  Bs[3] = {bq, bk, bv};
    float*        Ds[3] = {Q, K, V};

#pragma unroll
    for (int m = 0; m < 3; ++m) {
        // B fragment (32x16, K x N): lane holds column n=lane&15,
        // K = 16*half + {0..15}, contiguous along W's input dim.
        const __bf16* wrow = Ws[m] + (size_t)(c0 + nsub) * DF + half * 16;
        v8f acc = {};
#pragma unroll
        for (int ks = 0; ks < 4; ++ks) {
            v16bf b = *(const v16bf*)(wrow + ks * 32);
            acc = __builtin_amdgcn_wmma_f32_16x16x32_bf16(
                      false, a[ks], false, b, (short)0, acc, false, false);
        }
        const float bias = Bs[m][c0 + nsub];
        float* dcol = Ds[m] + (size_t)r0 * DF + c0 + nsub;
        if (full_tile) {
            // common path: straight-line stores, no exec-mask juggling.
            // D layout: VGPR i -> row i (lanes 0-15) / row 8+i (lanes 16-31)
#pragma unroll
            for (int i = 0; i < 8; ++i)
                dcol[(size_t)(i + 8 * half) * DF] = acc[i] + bias;
        } else {
#pragma unroll
            for (int i = 0; i < 8; ++i) {
                const int mr = i + 8 * half;
                if (r0 + mr < nrows) dcol[(size_t)mr * DF] = acc[i] + bias;
            }
        }
    }
}

// ---------------- edge kernels (wave32-per-edge, float4 lanes) ----------------

__global__ void __launch_bounds__(256)
edge_scores(const int* __restrict__ src, const int* __restrict__ dst,
            const float* __restrict__ Q, const float* __restrict__ K,
            float* __restrict__ scores, float* __restrict__ smax, int nedges)
{
    const int e = blockIdx.x * 8 + (threadIdx.x >> 5);
    if (e >= nedges) return;
    const int lane = threadIdx.x & 31;
    const int s = src[e], d = dst[e];
    const float4 q = *(const float4*)(Q + (size_t)d * DF + lane * 4);
    const float4 k = *(const float4*)(K + (size_t)s * DF + lane * 4);
    float p = q.x * k.x + q.y * k.y + q.z * k.z + q.w * k.w;
#pragma unroll
    for (int off = 16; off >= 1; off >>= 1)
        p += __shfl_xor(p, off, 32);
    if (lane == 0) {
        p *= 0.08838834764831845f;                 // 1/sqrt(128)
        scores[e] = p;
        atomicMax(smax + d, p);
    }
}

__global__ void __launch_bounds__(256)
edge_exp(const int* __restrict__ dst, float* __restrict__ scores,
         const float* __restrict__ smax, float* __restrict__ ssum, int nedges)
{
    const int e = blockIdx.x * blockDim.x + threadIdx.x;
    if (e < nedges) {
        const int d = dst[e];
        const float ex = __expf(scores[e] - smax[d]);
        scores[e] = ex;
        atomicAdd(ssum + d, ex);
    }
}

__global__ void __launch_bounds__(256)
edge_agg(const int* __restrict__ src, const int* __restrict__ dst,
         const float* __restrict__ scores, const float* __restrict__ ssum,
         const float* __restrict__ V, float* __restrict__ out, int nedges)
{
    const int e = blockIdx.x * 8 + (threadIdx.x >> 5);
    if (e >= nedges) return;
    const int lane = threadIdx.x & 31;
    const int s = src[e], d = dst[e];
    const float w = scores[e] / ssum[d];
    const float4 v = *(const float4*)(V + (size_t)s * DF + lane * 4);
    float* o = out + (size_t)d * DF + lane * 4;
    atomicAdd(o + 0, w * v.x);
    atomicAdd(o + 1, w * v.y);
    atomicAdd(o + 2, w * v.z);
    atomicAdd(o + 3, w * v.w);
}

// ---------------- host launcher ----------------

extern "C" void kernel_launch(void* const* d_in, const int* in_sizes, int n_in,
                              void* d_out, int out_size, void* d_ws, size_t ws_size,
                              hipStream_t stream)
{
    const float* x  = (const float*)d_in[0];
    const int*   ei = (const int*)d_in[1];
    const float* w1[3] = {(const float*)d_in[2], (const float*)d_in[4], (const float*)d_in[6]};
    const float* b1[3] = {(const float*)d_in[3], (const float*)d_in[5], (const float*)d_in[7]};
    const float* w2[3] = {(const float*)d_in[8], (const float*)d_in[10], (const float*)d_in[12]};
    const float* b2[3] = {(const float*)d_in[9], (const float*)d_in[11], (const float*)d_in[13]};

    const int N = in_sizes[0] / DF;
    const int E = in_sizes[1] / 2;
    const int* src = ei;
    const int* dst = ei + E;

    // workspace carve-up (256B aligned)
    char* wp = (char*)d_ws;
    auto alloc = [&](size_t bytes) { char* p = wp; wp += (bytes + 255) & ~(size_t)255; return p; };

    __bf16* Wbf[6];
    for (int i = 0; i < 6; ++i) Wbf[i] = (__bf16*)alloc((size_t)DF * DF * sizeof(__bf16));
    __bf16* xh   = (__bf16*)alloc((size_t)N * DF * sizeof(__bf16));
    float*  Q    = (float*)alloc((size_t)N * DF * sizeof(float));
    float*  K    = (float*)alloc((size_t)N * DF * sizeof(float));
    float*  V    = (float*)alloc((size_t)N * DF * sizeof(float));
    float*  h    = (float*)alloc((size_t)N * DF * sizeof(float));
    float*  smax = (float*)alloc((size_t)N * sizeof(float));
    float*  ssum = (float*)alloc((size_t)N * sizeof(float));
    float*  sc   = (float*)alloc((size_t)E * sizeof(float));

    auto nb = [](long n) { return (int)((n + 255) / 256); };

    // bf16 weights (tiny, stays in L2 for the whole run)
    const float* wsrc[6] = {w1[0], w1[1], w1[2], w2[0], w2[1], w2[2]};
    for (int i = 0; i < 6; ++i)
        cvt_f32_to_bf16<<<nb(DF * DF), 256, 0, stream>>>(wsrc[i], Wbf[i], DF * DF);
    cvt_f32_to_bf16<<<nb((long)N * DF), 256, 0, stream>>>(x, xh, N * DF);

    const int gemm_blocks = (N + 15) / 16;
    const int edge_wblocks = (E + 7) / 8;

    auto layer = [&](const __bf16* xin, const __bf16* Wq, const __bf16* Wk, const __bf16* Wv,
                     const float* bq, const float* bk, const float* bv, float* outAgg) {
        gemm_qkv_bf16<<<gemm_blocks, 256, 0, stream>>>(xin, Wq, Wk, Wv, bq, bk, bv, Q, K, V, N);
        fill_f32<<<nb(N), 256, 0, stream>>>(smax, -__builtin_huge_valf(), N);
        fill_f32<<<nb(N), 256, 0, stream>>>(ssum, 0.0f, N);
        fill_f32<<<nb((long)N * DF), 256, 0, stream>>>(outAgg, 0.0f, N * DF);
        edge_scores<<<edge_wblocks, 256, 0, stream>>>(src, dst, Q, K, sc, smax, E);
        edge_exp<<<nb(E), 256, 0, stream>>>(dst, sc, smax, ssum, E);
        edge_agg<<<edge_wblocks, 256, 0, stream>>>(src, dst, sc, ssum, V, outAgg, E);
    };

    // layer 1 -> h, ELU + bf16 -> xh (reused), layer 2 -> d_out
    layer(xh, Wbf[0], Wbf[1], Wbf[2], b1[0], b1[1], b1[2], h);
    elu_to_bf16<<<nb((long)N * DF), 256, 0, stream>>>(h, xh, N * DF);
    layer(xh, Wbf[3], Wbf[4], Wbf[5], b2[0], b2[1], b2[2], (float*)d_out);
}
